// SelfAttention_17154099380388
// MI455X (gfx1250) — matrix operand order repaired and checked by reading, not compiled, not run on hardware
//
#include <hip/hip_runtime.h>
#include <hip/hip_bf16.h>
#include <math.h>

typedef __attribute__((ext_vector_type(16))) _Float16 v16h;
typedef __attribute__((ext_vector_type(8)))  _Float16 v8h;
typedef __attribute__((ext_vector_type(2)))  _Float16 h2;
typedef __attribute__((ext_vector_type(8)))  float    v8f;

#define BATCH 4
#define CDIM  64
#define NPOS  4096
#define CQ    8

// ---- CDNA5 async global->LDS copy (ASYNCcnt path), guarded ----------------
#if defined(__has_builtin)
#if __has_builtin(__builtin_amdgcn_global_load_async_to_lds_b128) && \
    __has_builtin(__builtin_amdgcn_s_wait_asynccnt)
#define USE_ASYNC_LDS 1
#endif
#if __has_builtin(__builtin_amdgcn_tensor_load_to_lds) && \
    __has_builtin(__builtin_amdgcn_s_wait_tensorcnt)
#define USE_TDM 1
#endif
#endif
#ifndef USE_ASYNC_LDS
#define USE_ASYNC_LDS 0
#endif
#ifndef USE_TDM
#define USE_TDM 0
#endif

#if USE_ASYNC_LDS
typedef int v4i_ __attribute__((vector_size(16)));
typedef __attribute__((address_space(1))) v4i_ GV4;   // global int4
typedef __attribute__((address_space(3))) v4i_ LV4;   // LDS int4
__device__ __forceinline__ void async_copy16(const void* gp, void* lp) {
    GV4* g = (GV4*)(unsigned long long)(uintptr_t)gp;
    LV4* l = (LV4*)(unsigned int)(uintptr_t)lp;
    __builtin_amdgcn_global_load_async_to_lds_b128(g, l, 0, 0);
}
__device__ __forceinline__ void async_wait0() {
    __builtin_amdgcn_s_wait_asynccnt(0);
}
#else
__device__ __forceinline__ void async_wait0() {}
#endif

#if USE_TDM
typedef unsigned int u32x4 __attribute__((ext_vector_type(4)));
typedef int          i32x8 __attribute__((ext_vector_type(8)));
typedef int          i32x4 __attribute__((ext_vector_type(4)));
// 2D TDM tile load: tile_d0 8-byte units per row, tile_d1 rows,
// global row stride / tensor_d0 in 8-byte units. (ISA 8.3/8.4 D# packing.)
__device__ __forceinline__ void tdm_load_2d(const void* gp, void* lp,
                                            int tile_d0, int tile_d1,
                                            int tensor_d0, int row_stride) {
    unsigned long long ga = (unsigned long long)(uintptr_t)gp;
    u32x4 g0;
    g0[0] = 1u;                                        // count=1, user mode
    g0[1] = (unsigned)(uintptr_t)lp;                   // lds_addr
    g0[2] = (unsigned)(ga & 0xFFFFFFFFull);            // global_addr[31:0]
    g0[3] = (unsigned)((ga >> 32) & 0x01FFFFFFull)     // global_addr[56:32]
          | (2u << 30);                                // type=2 ("image")
    i32x8 g1;
    g1[0] = 3 << 16;                                   // data_size=8B
    g1[1] = (tensor_d0 & 0xFFFF) << 16;                // tensor_dim0[15:0]
    g1[2] = ((tensor_d0 >> 16) & 0xFFFF)
          | ((tile_d1 /*tensor_dim1*/ & 0xFFFF) << 16);
    g1[3] = (tile_d0 & 0xFFFF) << 16;                  // tile_dim0
    g1[4] = tile_d1 & 0xFFFF;                          // tile_dim1 (tile_dim2=0)
    g1[5] = row_stride;                                // tensor_dim0_stride[31:0]
    g1[6] = 0;
    g1[7] = 0;
    i32x4 z4 = {0, 0, 0, 0};
    i32x8 z8 = {0, 0, 0, 0, 0, 0, 0, 0};
    __builtin_amdgcn_tensor_load_to_lds(g0, g1, z4, z4, z8, 0);
}
__device__ __forceinline__ void tensor_wait0() {
    __builtin_amdgcn_s_wait_tensorcnt(0);
}
#endif

// ---- DPP16 butterfly max over each 16-lane row (pure VALU) ----------------
template <int CTRL>
__device__ __forceinline__ float dpp_movf(float v) {
    return __int_as_float(__builtin_amdgcn_update_dpp(
        0, __float_as_int(v), CTRL, 0xf, 0xf, true));
}
__device__ __forceinline__ float group16_max(float v) {
    v = fmaxf(v, dpp_movf<0xB1>(v));   // quad_perm [1,0,3,2]  : xor 1
    v = fmaxf(v, dpp_movf<0x4E>(v));   // quad_perm [2,3,0,1]  : xor 2
    v = fmaxf(v, dpp_movf<0x141>(v));  // row_half_mirror      : xor 4
    v = fmaxf(v, dpp_movf<0x140>(v));  // row_mirror           : xor 8
    return v;
}

// ---------------------------------------------------------------------------
// Kernel 1: QKV projection (1x1 conv == channel-mixing GEMM), fp32 -> f16.
// Q:  [B][N][8]  row-major per query  (A fragments)
// Kt: [B][N][8]  row-major per key    (B fragments: contiguous 16B per key)
// Vt: [B][N][64] key-major transposed V
// ---------------------------------------------------------------------------
__global__ __launch_bounds__(256) void qkv_kernel(
    const float* __restrict__ x,
    const float* __restrict__ Wq, const float* __restrict__ bq,
    const float* __restrict__ Wk, const float* __restrict__ bk,
    const float* __restrict__ Wv, const float* __restrict__ bv,
    _Float16* __restrict__ Qh, _Float16* __restrict__ Kt,
    _Float16* __restrict__ Vt)
{
    __shared__ float xs[CDIM][68];
    __shared__ float wq[CQ][CDIM];
    __shared__ float wk[CQ][CDIM];
    __shared__ float wv[CDIM][CDIM];
    __shared__ float bqs[CQ], bks[CQ], bvs[CDIM];

    const int b  = blockIdx.x >> 6;
    const int n0 = (blockIdx.x & 63) * 64;
    const int t  = threadIdx.x;

#if USE_ASYNC_LDS
    #pragma unroll
    for (int i = 0; i < 4; ++i) {
        int idx = t + i * 256;
        int c = idx >> 4, q4 = (idx & 15) * 4;
        async_copy16(&x[((size_t)(b * CDIM + c)) * NPOS + n0 + q4], &xs[c][q4]);
    }
    #pragma unroll
    for (int i = 0; i < 4; ++i) {
        int idx = (t + i * 256) * 4;
        async_copy16(&Wv[idx], &((float*)wv)[idx]);
    }
    if (t < 128) {
        async_copy16(&Wq[t * 4], &((float*)wq)[t * 4]);
        async_copy16(&Wk[t * 4], &((float*)wk)[t * 4]);
    }
#else
    #pragma unroll
    for (int i = 0; i < 16; ++i) {
        int idx = t + i * 256;
        int c = idx >> 6, n = idx & 63;
        xs[c][n] = x[((size_t)(b * CDIM + c)) * NPOS + n0 + n];
    }
    #pragma unroll
    for (int i = 0; i < 2; ++i) {
        int idx = t + i * 256;
        ((float*)wq)[idx] = Wq[idx];
        ((float*)wk)[idx] = Wk[idx];
    }
    #pragma unroll
    for (int i = 0; i < 16; ++i) {
        int idx = t + i * 256;
        ((float*)wv)[idx] = Wv[idx];
    }
#endif
    if (t < CQ)   { bqs[t] = bq[t]; bks[t] = bk[t]; }
    if (t < CDIM) { bvs[t] = bv[t]; }
    async_wait0();
    __syncthreads();

    const int g  = t >> 6;
    const int tn = t & 63;

    #pragma unroll
    for (int cc = 0; cc < 16; ++cc) {
        int c = g * 16 + cc;
        float acc = bvs[c];
        #pragma unroll 8
        for (int k = 0; k < CDIM; ++k) acc += wv[c][k] * xs[k][tn];
        Vt[((size_t)b * NPOS + n0 + tn) * CDIM + c] = (_Float16)acc;
    }
    if (g == 0) {
        #pragma unroll
        for (int cq = 0; cq < CQ; ++cq) {
            float acc = bqs[cq];
            #pragma unroll 8
            for (int k = 0; k < CDIM; ++k) acc += wq[cq][k] * xs[k][tn];
            Qh[((size_t)b * NPOS + n0 + tn) * CQ + cq] = (_Float16)acc;
        }
    } else if (g == 1) {
        #pragma unroll
        for (int cq = 0; cq < CQ; ++cq) {
            float acc = bks[cq];
            #pragma unroll 8
            for (int k = 0; k < CDIM; ++k) acc += wk[cq][k] * xs[k][tn];
            Kt[((size_t)b * NPOS + n0 + tn) * CQ + cq] = (_Float16)acc;  // transposed
        }
    }
}

// ---------------------------------------------------------------------------
// Kernel 2: fused flash-attention, double-buffered async pipeline.
// ---------------------------------------------------------------------------
#define KT 64
#define NT (NPOS / KT)
#define VPAD 80

__global__ __launch_bounds__(256) void attn_kernel(
    const _Float16* __restrict__ Qh, const _Float16* __restrict__ Kt,
    const _Float16* __restrict__ Vt, const float* __restrict__ x,
    const float* __restrict__ gamma, float* __restrict__ out)
{
    __shared__ __align__(16) _Float16 Ks[2][KT][CQ];   // [key n][cq], 16B rows
    __shared__ __align__(16) _Float16 Vs[2][CDIM][VPAD];
    __shared__ __align__(16) _Float16 Ps[8][16][VPAD];

    const int b    = blockIdx.y;
    const int q0   = blockIdx.x * 128;
    const int t    = threadIdx.x;
    const int wave = t >> 5;
    const int lane = t & 31;
    const int half = lane >> 4;
    const int l16  = lane & 15;

    // V staging assignment: one (key-pair, 8-channel group) slot per thread
    const int np = (t >> 3) * 2;       // 0,2,...,62
    const int cg = (t & 7) * 8;        // 0,8,...,56
    v8h vreg0, vreg1;

    // Q fragment (A 16x32 f16, K 8..31 zero)
    const int qrow = q0 + wave * 16 + l16;
    v16h aq;
    #pragma unroll
    for (int e = 0; e < 16; ++e) aq[e] = (_Float16)0.0f;
    if (half == 0) {
        const _Float16* qp = Qh + ((size_t)b * NPOS + qrow) * CQ;
        #pragma unroll
        for (int e = 0; e < CQ; ++e) aq[e] = qp[e];
    }
    // all-ones B fragment: row sums via the matrix core
    v16h ones;
    #pragma unroll
    for (int e = 0; e < 16; ++e) ones[e] = (_Float16)1.0f;

    v8f og[4], og_l;
    float m[8];
    #pragma unroll
    for (int r = 0; r < 8; ++r) { m[r] = -3.0e38f; og_l[r] = 0.0f; }
    #pragma unroll
    for (int ch = 0; ch < 4; ++ch)
        #pragma unroll
        for (int r = 0; r < 8; ++r) og[ch][r] = 0.0f;

    // ---- staging helpers -------------------------------------------------
    auto load_k = [&](int j0, int buf) {
        // K tile = contiguous 1KB block Kt[b][j0..j0+63][0..7]
        const _Float16* gk = Kt + ((size_t)b * NPOS + j0) * CQ;
#if USE_TDM
        if (wave == 0) {
            // contiguous copy: 128 x 8B, one row
            tdm_load_2d(gk, &Ks[buf][0][0],
                        /*tile_d0=*/128, /*tile_d1=*/1,
                        /*tensor_d0=*/128, /*row_stride=*/128);
        }
#elif USE_ASYNC_LDS
        if (t < 64) {
            async_copy16(gk + t * 8, &Ks[buf][0][0] + t * 8);
        }
#else
        {
            _Float16* ksf = &Ks[buf][0][0];
            ksf[t * 2]     = gk[t * 2];
            ksf[t * 2 + 1] = gk[t * 2 + 1];
        }
#endif
    };
    auto load_v = [&](int j0) {   // global -> regs (latency hidden by compute)
        const _Float16* vp = Vt + ((size_t)b * NPOS + j0 + np) * CDIM + cg;
        vreg0 = *(const v8h*)vp;
        vreg1 = *(const v8h*)(vp + CDIM);
    };
    auto store_v = [&](int buf) { // regs -> LDS transposed, packed b32 stores
        #pragma unroll
        for (int c = 0; c < 8; ++c) {
            h2 pr; pr[0] = vreg0[c]; pr[1] = vreg1[c];
            *(h2*)&Vs[buf][cg + c][np] = pr;
        }
    };
    auto wait_stage = [&]() {
#if USE_TDM
        if (wave == 0) tensor_wait0();
#endif
        async_wait0();
    };

    // ---- prologue: stage tile 0 -----------------------------------------
    load_k(0, 0);
    load_v(0);
    store_v(0);
    wait_stage();
    __syncthreads();

    for (int jt = 0; jt < NT; ++jt) {
        const int cur  = jt & 1;
        const int j0   = jt * KT;
        const bool more = (jt + 1) < NT;

        // issue next tile's loads early: TDM/async K + global V into regs
        if (more) {
            load_k(j0 + KT, cur ^ 1);
            load_v(j0 + KT);
        }

        // ---- S = Q K : 4 tiles of 16x16 ----------------------------------
        v8f s[4];
        #pragma unroll
        for (int tt = 0; tt < 4; ++tt) {
            v16h bk;
            #pragma unroll
            for (int e = 0; e < 16; ++e) bk[e] = (_Float16)0.0f;
            int ncol = tt * 16 + l16;
            if (half == 0) {   // one contiguous 16B LDS read per lane
                v8h kr = *(const v8h*)&Ks[cur][ncol][0];
                #pragma unroll
                for (int e = 0; e < CQ; ++e) bk[e] = kr[e];
            }
            v8f zero;
            #pragma unroll
            for (int r = 0; r < 8; ++r) zero[r] = 0.0f;
            s[tt] = __builtin_amdgcn_wmma_f32_16x16x32_f16(
                false, aq, false, bk, (short)0, zero, false, false);
        }

        // ---- online softmax: DPP16 max, exp; sums via WMMA ---------------
        #pragma unroll
        for (int r = 0; r < 8; ++r) {
            float v = fmaxf(fmaxf(s[0][r], s[1][r]), fmaxf(s[2][r], s[3][r]));
            v = group16_max(v);
            float mn    = fmaxf(m[r], v);
            float alpha = __expf(m[r] - mn);
            m[r] = mn;
            #pragma unroll
            for (int tt = 0; tt < 4; ++tt)
                s[tt][r] = __expf(s[tt][r] - mn);
            og_l[r] *= alpha;
            #pragma unroll
            for (int ch = 0; ch < 4; ++ch) og[ch][r] *= alpha;
        }

        // ---- spill P to per-wave LDS (C layout -> A layout regroup) ------
        #pragma unroll
        for (int tt = 0; tt < 4; ++tt)
            #pragma unroll
            for (int r = 0; r < 8; ++r)
                Ps[wave][r + 8 * half][tt * 16 + l16] = (_Float16)s[tt][r];

        // ---- O += P Vt ; l += P 1 ----------------------------------------
        #pragma unroll
        for (int kk = 0; kk < 2; ++kk) {
            v16h ap;
            const _Float16* prow = &Ps[wave][l16][0];
            #pragma unroll
            for (int e = 0; e < 16; ++e) {
                int k = kk * 32 + half * 16 + e;
                ap[e] = prow[k];
            }
            #pragma unroll
            for (int ch = 0; ch < 4; ++ch) {
                v16h bvf;
                int c = ch * 16 + l16;
                const _Float16* vrow = &Vs[cur][c][kk * 32 + half * 16];
                #pragma unroll
                for (int e = 0; e < 16; ++e) bvf[e] = vrow[e];
                og[ch] = __builtin_amdgcn_wmma_f32_16x16x32_f16(
                    false, ap, false, bvf, (short)0, og[ch], false, false);
            }
            og_l = __builtin_amdgcn_wmma_f32_16x16x32_f16(
                false, ap, false, ones, (short)0, og_l, false, false);
        }

        // ---- finish staging of next tile, then hand buffers over ---------
        if (more) {
            store_v(cur ^ 1);
            wait_stage();
        }
        __syncthreads();
    }

    // ---- epilogue: normalize, gamma*out + x ------------------------------
    const float g = gamma[0];
    #pragma unroll
    for (int ch = 0; ch < 4; ++ch) {
        #pragma unroll
        for (int r = 0; r < 8; ++r) {
            int n = q0 + wave * 16 + r + 8 * half;
            int c = ch * 16 + l16;
            size_t idx = ((size_t)(b * CDIM + c)) * NPOS + n;
            float val = og[ch][r] / og_l[r];
            out[idx] = g * val + x[idx];
        }
    }
}

// ---------------------------------------------------------------------------
extern "C" void kernel_launch(void* const* d_in, const int* in_sizes, int n_in,
                              void* d_out, int out_size, void* d_ws, size_t ws_size,
                              hipStream_t stream) {
    const float* x     = (const float*)d_in[0];
    const float* Wq    = (const float*)d_in[1];
    const float* bq    = (const float*)d_in[2];
    const float* Wk    = (const float*)d_in[3];
    const float* bk    = (const float*)d_in[4];
    const float* Wv    = (const float*)d_in[5];
    const float* bv    = (const float*)d_in[6];
    const float* gamma = (const float*)d_in[7];
    float* out = (float*)d_out;

    _Float16* Qh = (_Float16*)d_ws;                       // [4][4096][8]
    _Float16* Kt = Qh + (size_t)BATCH * NPOS * CQ;        // [4][4096][8]
    _Float16* Vt = Kt + (size_t)BATCH * NPOS * CQ;        // [4][4096][64]

    qkv_kernel<<<BATCH * (NPOS / 64), 256, 0, stream>>>(
        x, Wq, bq, Wk, bk, Wv, bv, Qh, Kt, Vt);

    dim3 grid(NPOS / 128, BATCH);
    attn_kernel<<<grid, 256, 0, stream>>>(Qh, Kt, Vt, x, gamma, out);
}
